// LinearSpatCosSimTempConvNet_41686952575615
// MI455X (gfx1250) — matrix operand order repaired and checked by reading, not compiled, not run on hardware
//
#include <hip/hip_runtime.h>

typedef _Float16 v16h __attribute__((ext_vector_type(16)));
typedef float    v8f  __attribute__((ext_vector_type(8)));

#define Bsz   32
#define Cch   64
#define Tlen  2048
#define Fft   256
#define Kk    64
#define TOUT  (Tlen - Kk + 1)      // 1985
#define CHUNK 512
#define SCOLS (CHUNK + Kk)         // 576 columns of spat_out kept in LDS
#define SSTRIDE (SCOLS + 4)        // pad to kill bank conflicts (8*580 % 64 == 32)

__global__ void zero_out_kernel(float* __restrict__ out) {
    if (threadIdx.x < Bsz) out[threadIdx.x] = 0.0f;
}

__global__ __launch_bounds__(256)
void fused_spatcos_kernel(const float* __restrict__ x,        // (B, C, T)
                          const float* __restrict__ conv_w,   // (F, K)
                          const float* __restrict__ spat_w,   // (F, C)
                          const float* __restrict__ weight,   // (F,)
                          const float* __restrict__ bias,     // (F,)
                          float* __restrict__ out)            // (B,)
{
    __shared__ float sh[16 * SSTRIDE];   // 16 filters x 580 cols of spat_out
    __shared__ float red[256];

    const int tid  = threadIdx.x;
    const int lane = tid & 31;
    const int hi   = lane >> 4;          // 0: lanes 0-15, 1: lanes 16-31
    const int f0   = blockIdx.x * 16;    // filter tile base
    const int b    = blockIdx.y;         // batch

    // ---------------- A matrix: spat_w tile (16 x 64), f16, contiguous K order ----------------
    // lane m = lane&15 holds row M=m; halves 0..15 -> K = hi*16 + 0..15
    // (contiguous, matching B), two K-chunks.
    const int   am   = lane & 15;
    const float* wrow = spat_w + (size_t)(f0 + am) * Cch;
    v16h a0, a1;
#pragma unroll
    for (int h = 0; h < 16; ++h) {
        int kk = hi * 16 + h;                          // K within the 32-chunk
        a0[h] = (_Float16)wrow[kk];                    // K = 0..31
        a1[h] = (_Float16)wrow[32 + kk];               // K = 32..63
    }

    // ---------------- conv weights for this thread's filter (registers) ----------------
    const int f = tid >> 4;              // 0..15 (filter within tile)
    const int j = tid & 15;              // position lane within filter
    const float* cwrow = conv_w + (size_t)(f0 + f) * Kk;
    float wreg[Kk];
    float nsq = 0.0f;
#pragma unroll
    for (int k = 0; k < Kk; ++k) {
        wreg[k] = cwrow[k];
        nsq += wreg[k] * wreg[k];
    }
    const float cnorm = 8.0f * rsqrtf(nsq);  // sqrt(K)/||w_f||, K = 64

    const float* xb = x + (size_t)b * Cch * Tlen;
    const int koff = hi * 16;            // B-matrix K offset for lanes 16-31
    const int bn   = lane & 15;          // B-matrix column N
    const int wave = tid >> 5;           // 8 waves per block

    float thread_abs = 0.0f;

    for (int chunk = 0; chunk < 4; ++chunk) {
        const int tc = chunk * CHUNK;

        // ---------- Phase A: WMMA S(16 x 576) = A(16x64) @ X[b](64x576) into LDS ----------
        for (int tile = wave; tile < SCOLS / 16; tile += 8) {
            const int t0 = tc + tile * 16;
            if (t0 >= Tlen) break;
            // B matrix (32 x 16) f16, contiguous K order: lane holds column N=bn,
            // half h -> K = koff + h
            const float* xcol = xb + (size_t)koff * Tlen + t0 + bn;
            v16h b0, b1;
#pragma unroll
            for (int h = 0; h < 16; ++h) {
                b0[h] = (_Float16)xcol[(size_t)h * Tlen];                 // K = 0..31
                b1[h] = (_Float16)xcol[(size_t)(32 + h) * Tlen - (size_t)koff * Tlen
                                       + (size_t)koff * Tlen];            // K = 32..63
            }
            // (the b1 index above is just xcol + 32 rows; keep it simple:)
#pragma unroll
            for (int h = 0; h < 16; ++h) {
                b1[h] = (_Float16)xcol[(size_t)(32 + h) * Tlen];
            }

            v8f c = {};
            c = __builtin_amdgcn_wmma_f32_16x16x32_f16(false, a0, false, b0,
                                                       (short)0, c, false, false);
            c = __builtin_amdgcn_wmma_f32_16x16x32_f16(false, a1, false, b1,
                                                       (short)0, c, false, false);

            // D layout: VGPR r -> row M = r + 8*hi, col N = lane&15
            const int coll = (t0 - tc) + bn;
            const int mrow = hi * 8;
#pragma unroll
            for (int r = 0; r < 8; ++r)
                sh[(r + mrow) * SSTRIDE + coll] = c[r];
        }
        __syncthreads();

        // ---------- Phase B: depthwise conv + window sum-of-squares from LDS ----------
        const int tend = (tc + CHUNK < TOUT) ? (tc + CHUNK) : TOUT;
        const float* srow = &sh[f * SSTRIDE];
        for (int t = tc + j; t < tend; t += 16) {
            const int tl = t - tc;
            float accC = 0.0f, accQ = 0.0f;
#pragma unroll
            for (int k = 0; k < Kk; ++k) {
                float s = srow[tl + k];
                accC += s * wreg[k];
                accQ += s * s;
            }
            thread_abs += fabsf(accC) * rsqrtf(accQ);
        }
        __syncthreads();   // S reused next chunk
    }

    // ---------- finalize: weight * mean(|cos|) per filter, reduce, atomicAdd ----------
    red[tid] = thread_abs * cnorm * weight[f0 + f] * (1.0f / (float)TOUT);
    __syncthreads();
#pragma unroll
    for (int s = 128; s > 0; s >>= 1) {
        if (tid < s) red[tid] += red[tid + s];
        __syncthreads();
    }
    if (tid == 0) {
        float bs = 0.0f;
#pragma unroll
        for (int ff = 0; ff < 16; ++ff) bs += bias[f0 + ff];
        atomicAdd(&out[b], red[0] + bs);
    }
}

extern "C" void kernel_launch(void* const* d_in, const int* in_sizes, int n_in,
                              void* d_out, int out_size, void* d_ws, size_t ws_size,
                              hipStream_t stream) {
    (void)in_sizes; (void)n_in; (void)d_ws; (void)ws_size; (void)out_size;
    const float* x      = (const float*)d_in[0];
    const float* conv_w = (const float*)d_in[1];
    const float* spat_w = (const float*)d_in[2];
    const float* weight = (const float*)d_in[3];
    const float* bias   = (const float*)d_in[4];
    float* out = (float*)d_out;

    zero_out_kernel<<<1, 32, 0, stream>>>(out);
    dim3 grid(Fft / 16, Bsz);   // 16 filter-tiles x 32 batches = 512 workgroups
    fused_spatcos_kernel<<<grid, 256, 0, stream>>>(x, conv_w, spat_w, weight, bias, out);
}
